// Quantize_43293270344327
// MI455X (gfx1250) — matrix operand order repaired and checked by reading, not compiled, not run on hardware
//
#include <hip/hip_runtime.h>
#include <hip/hip_bf16.h>
#include <cstdint>

// Problem constants (from reference): T=2048, B=16, C=512, N=4096
#define T_DIM 2048
#define B_DIM 16
#define C_DIM 512
#define N_DIM 4096
#define M_DIM (T_DIM * B_DIM)        // 32768 rows
#define QELEMS (M_DIM * C_DIM)       // 16777216 quantize elements
#define NT_TILES (N_DIM / 16)        // 256 N-tiles of 16 cols
#define KT_TILES (C_DIM / 32)        // 16 K-tiles of 32

typedef __attribute__((ext_vector_type(16))) _Float16 v16h;
typedef __attribute__((ext_vector_type(8)))  float    v8f;

// -------- workspace layout (bytes) --------
#define BPH_OFF   0ull                                          // B hi f16 frags: 4 MiB
#define BPL_OFF   (BPH_OFF + (size_t)N_DIM * C_DIM * 2)         // B lo f16 frags: 4 MiB
#define EMBT_OFF  (BPL_OFF + (size_t)N_DIM * C_DIM * 2)         // embed^T f32:    8 MiB
#define ENORM_OFF (EMBT_OFF + (size_t)N_DIM * C_DIM * 4)        // ||e||^2:       16 KiB
#define IDX_OFF   (ENORM_OFF + (size_t)N_DIM * 4)               // argmin idx:   128 KiB
#define CNT_OFF   (IDX_OFF + (size_t)M_DIM * 4)                 // counts:        16 KiB
#define DIFF_OFF  (CNT_OFF + (size_t)N_DIM * 4)                 // diff accum:      4 B

// ------------------------------------------------------------------
// zero per-call accumulators (harness does not re-zero between replays)
__global__ void k_init(float* __restrict__ counts, float* __restrict__ diffsum) {
    int gid = blockIdx.x * 256 + threadIdx.x;
    if (gid < N_DIM) counts[gid] = 0.0f;
    if (gid == 0) *diffsum = 0.0f;
}

// ------------------------------------------------------------------
// Pre-swizzle embed [C,N] into WMMA B-fragment lane order, split f32 -> f16 hi/lo.
// B layout for V_WMMA_F32_16X16X32_F16 (32x16 KxN): lane n (0..15) holds col n,
// K = h + (h>=8?8:0); lane n+16 holds col n, K = h + (h>=8?8:0) + 8.
// Packed index: ((nt*16 + kt)*32 + lane)*16 + h
__global__ void k_pack_embed(const float* __restrict__ embed,
                             _Float16* __restrict__ bph, _Float16* __restrict__ bpl) {
    int gid = blockIdx.x * 256 + threadIdx.x;   // < N_DIM*C_DIM
    int h    = gid & 15;
    int lane = (gid >> 4) & 31;
    int kt   = (gid >> 9) & 15;
    int nt   = gid >> 13;
    int n = nt * 16 + (lane & 15);
    int k = kt * 32 + h + ((h & 8) ? 8 : 0) + ((lane & 16) ? 8 : 0);
    float v = embed[(size_t)k * N_DIM + n];
    _Float16 hi = (_Float16)v;
    bph[gid] = hi;
    bpl[gid] = (_Float16)(v - (float)hi);
}

// embed [C,N] -> embedT [N,C] so the per-row gather in pass 2 is contiguous
__global__ void k_transpose(const float* __restrict__ embed, float* __restrict__ embedT) {
    int gid = blockIdx.x * 256 + threadIdx.x;   // < N_DIM*C_DIM
    int n = gid >> 9;        // /C_DIM
    int k = gid & (C_DIM - 1);
    embedT[gid] = embed[(size_t)k * N_DIM + n];
}

// ||e_n||^2 per code (coalesced: fixed k, consecutive n)
__global__ void k_enorm(const float* __restrict__ embed, float* __restrict__ enorm) {
    int n = blockIdx.x * 256 + threadIdx.x;
    float s = 0.0f;
    for (int k = 0; k < C_DIM; ++k) {
        float e = embed[(size_t)k * N_DIM + n];
        s += e * e;
    }
    enorm[n] = s;
}

// ------------------------------------------------------------------
// Fused GEMM + argmin. Block = 256 threads = 8 waves; each wave owns 16 rows.
// Block M-tile = 128 rows. A panel staged in LDS as split-f16 WMMA A fragments
// (256 KiB of the 320 KiB/WGP). Score = ||e||^2 - 2*x.e  (||x||^2 is row-constant).
// x.e computed as hi*hi + hi*lo + lo*hi  (3x V_WMMA_F32_16X16X32_F16, ~f32 accurate).
__global__ __launch_bounds__(256) void k_gemm_argmin(
    const float* __restrict__ input, const _Float16* __restrict__ bph,
    const _Float16* __restrict__ bpl, const float* __restrict__ enorm,
    int* __restrict__ idxout) {
    __shared__ _Float16 sAh[8 * KT_TILES * 32 * 16];   // 128 KiB
    __shared__ _Float16 sAl[8 * KT_TILES * 32 * 16];   // 128 KiB
    const int tid  = threadIdx.x;
    const int lane = tid & 31;
    const int w    = tid >> 5;
    const int rowblk = blockIdx.x * 128;

    // Stage A: 128 rows x 512 K -> fragment order.
    // A layout (16x32 MxK): lanes 0-15: M=lane, K = h + (h>=8?8:0);
    //                       lanes 16-31: M=lane-16, K = h + (h>=8?8:0) + 8.
    for (int idx = tid; idx < 8 * KT_TILES * 32 * 16; idx += 256) {
        int h   = idx & 15;
        int ln  = (idx >> 4) & 31;
        int kt  = (idx >> 9) & 15;
        int m16 = idx >> 13;
        int r = m16 * 16 + (ln & 15);
        int k = kt * 32 + h + ((h & 8) ? 8 : 0) + ((ln & 16) ? 8 : 0);
        float v = input[(size_t)(rowblk + r) * C_DIM + k];
        _Float16 hi = (_Float16)v;
        sAh[idx] = hi;
        sAl[idx] = (_Float16)(v - (float)hi);
    }
    __syncthreads();

    float minv[8];
    int   mini[8];
#pragma unroll
    for (int r = 0; r < 8; ++r) { minv[r] = 3.4e38f; mini[r] = 0; }
    const int nlo = lane & 15;

    for (int nt = 0; nt < NT_TILES; nt += 2) {
        v8f acc0 = {0.f,0.f,0.f,0.f,0.f,0.f,0.f,0.f};
        v8f acc1 = {0.f,0.f,0.f,0.f,0.f,0.f,0.f,0.f};
#pragma unroll 4
        for (int kt = 0; kt < KT_TILES; ++kt) {
            const int aoff = ((w * KT_TILES + kt) * 32 + lane) * 16;
            v16h ah = *reinterpret_cast<const v16h*>(sAh + aoff);
            v16h al = *reinterpret_cast<const v16h*>(sAl + aoff);
            const int b0 = (((nt + 0) * KT_TILES + kt) * 32 + lane) * 16;
            const int b1 = (((nt + 1) * KT_TILES + kt) * 32 + lane) * 16;
            v16h bh0 = *reinterpret_cast<const v16h*>(bph + b0);
            v16h bl0 = *reinterpret_cast<const v16h*>(bpl + b0);
            v16h bh1 = *reinterpret_cast<const v16h*>(bph + b1);
            v16h bl1 = *reinterpret_cast<const v16h*>(bpl + b1);
            acc0 = __builtin_amdgcn_wmma_f32_16x16x32_f16(false, ah, false, bh0, (short)0, acc0, false, false);
            acc0 = __builtin_amdgcn_wmma_f32_16x16x32_f16(false, ah, false, bl0, (short)0, acc0, false, false);
            acc0 = __builtin_amdgcn_wmma_f32_16x16x32_f16(false, al, false, bh0, (short)0, acc0, false, false);
            acc1 = __builtin_amdgcn_wmma_f32_16x16x32_f16(false, ah, false, bh1, (short)0, acc1, false, false);
            acc1 = __builtin_amdgcn_wmma_f32_16x16x32_f16(false, ah, false, bl1, (short)0, acc1, false, false);
            acc1 = __builtin_amdgcn_wmma_f32_16x16x32_f16(false, al, false, bh1, (short)0, acc1, false, false);
        }
        // Epilogue: C layout — VGPR r: lanes0-15 = (M=r, N=lane), lanes16-31 = (M=8+r, N=lane-16)
        float en0 = enorm[(nt + 0) * 16 + nlo];
        float en1 = enorm[(nt + 1) * 16 + nlo];
        const int n0 = (nt + 0) * 16 + nlo;
        const int n1 = (nt + 1) * 16 + nlo;
#pragma unroll
        for (int r = 0; r < 8; ++r) {
            float s0 = en0 - 2.0f * acc0[r];
            float s1 = en1 - 2.0f * acc1[r];
            if (s0 < minv[r]) { minv[r] = s0; mini[r] = n0; }
            if (s1 < minv[r]) { minv[r] = s1; mini[r] = n1; }
        }
    }

    // Cross-lane argmin within each 16-lane half (rows r in lanes 0-15, rows 8+r in 16-31).
#pragma unroll
    for (int r = 0; r < 8; ++r) {
        float v = minv[r];
        int   i = mini[r];
#pragma unroll
        for (int m = 8; m >= 1; m >>= 1) {
            float ov = __shfl_xor(v, m, 32);
            int   oi = __shfl_xor(i, m, 32);
            if (ov < v || (ov == v && oi < i)) { v = ov; i = oi; }   // jnp.argmin tie-break: lowest index
        }
        if (nlo == 0) {
            int mrow = (lane < 16) ? r : (8 + r);
            idxout[rowblk + w * 16 + mrow] = i;
        }
    }
}

// ------------------------------------------------------------------
// Pass 2: gather chosen code, write quantize_st = x + (q - x) (reference expr),
// accumulate diff numerator and per-code counts; emit embed_ind as float.
__global__ __launch_bounds__(256) void k_finalize(
    const float* __restrict__ input, const unsigned char* __restrict__ mask,
    const float* __restrict__ embedT, const int* __restrict__ idxv,
    float* __restrict__ out, float* __restrict__ counts, float* __restrict__ diffsum) {
    __shared__ float red[256];
    size_t gid = (size_t)blockIdx.x * 256 + threadIdx.x;   // < QELEMS
    int row = (int)(gid >> 9);
    int k   = (int)(gid & (C_DIM - 1));
    int idx = idxv[row];
    float fm = mask[row] ? 1.0f : 0.0f;
    float q = embedT[(size_t)idx * C_DIM + k] * fm;
    float x = input[gid] * fm;
    float d = q - x;
    out[gid] = x + d;                                      // straight-through value
    if (k == 0) {
        out[(size_t)QELEMS + 1 + row] = (float)idx;        // embed_ind output slot
        atomicAdd(&counts[idx], fm);
    }
    red[threadIdx.x] = d * d;
    __syncthreads();
    for (int s = 128; s > 0; s >>= 1) {
        if (threadIdx.x < s) red[threadIdx.x] += red[threadIdx.x + s];
        __syncthreads();
    }
    if (threadIdx.x == 0) atomicAdd(diffsum, red[0]);
}

// Final scalars: diff mean and effective_units = 1 / sum((counts/wsum)^2)
__global__ __launch_bounds__(256) void k_scalars(const float* __restrict__ counts,
                                                 const float* __restrict__ diffsum,
                                                 float* __restrict__ out) {
    __shared__ float red[256];
    float s = 0.0f;
    for (int n = threadIdx.x; n < N_DIM; n += 256) s += counts[n];
    red[threadIdx.x] = s;
    __syncthreads();
    for (int st = 128; st > 0; st >>= 1) {
        if (threadIdx.x < st) red[threadIdx.x] += red[threadIdx.x + st];
        __syncthreads();
    }
    float wsum = red[0];
    __syncthreads();
    float s2 = 0.0f;
    for (int n = threadIdx.x; n < N_DIM; n += 256) {
        float mo = counts[n] / wsum;
        s2 += mo * mo;
    }
    red[threadIdx.x] = s2;
    __syncthreads();
    for (int st = 128; st > 0; st >>= 1) {
        if (threadIdx.x < st) red[threadIdx.x] += red[threadIdx.x + st];
        __syncthreads();
    }
    if (threadIdx.x == 0) {
        out[QELEMS] = (*diffsum) / (float)QELEMS;                 // diff
        out[(size_t)QELEMS + 1 + M_DIM] = 1.0f / red[0];          // effective_units
    }
}

// ------------------------------------------------------------------
extern "C" void kernel_launch(void* const* d_in, const int* in_sizes, int n_in,
                              void* d_out, int out_size, void* d_ws, size_t ws_size,
                              hipStream_t stream) {
    const float*         input = (const float*)d_in[0];          // [T,B,C] f32
    const unsigned char* mask  = (const unsigned char*)d_in[1];  // [T,B] bool
    const float*         embed = (const float*)d_in[2];          // [C,N] f32
    float* out = (float*)d_out;
    char*  ws  = (char*)d_ws;

    _Float16* bph    = (_Float16*)(ws + BPH_OFF);
    _Float16* bpl    = (_Float16*)(ws + BPL_OFF);
    float*    embedT = (float*)(ws + EMBT_OFF);
    float*    enorm  = (float*)(ws + ENORM_OFF);
    int*      idxv   = (int*)(ws + IDX_OFF);
    float*    counts = (float*)(ws + CNT_OFF);
    float*    diffsum= (float*)(ws + DIFF_OFF);

    k_init<<<16, 256, 0, stream>>>(counts, diffsum);
    k_pack_embed<<<(N_DIM * C_DIM) / 256, 256, 0, stream>>>(embed, bph, bpl);
    k_transpose<<<(N_DIM * C_DIM) / 256, 256, 0, stream>>>(embed, embedT);
    k_enorm<<<N_DIM / 256, 256, 0, stream>>>(embed, enorm);
    k_gemm_argmin<<<M_DIM / 128, 256, 0, stream>>>(input, bph, bpl, enorm, idxv);
    k_finalize<<<QELEMS / 256, 256, 0, stream>>>(input, mask, embedT, idxv, out, counts, diffsum);
    k_scalars<<<1, 256, 0, stream>>>(counts, diffsum, out);
}
